// MultiSpark_19997367730509
// MI455X (gfx1250) — matrix operand order repaired and checked by reading, not compiled, not run on hardware
//
#include <hip/hip_runtime.h>
#include <hip/hip_bf16.h>

#define NN  4096
#define KSP 32

typedef __attribute__((ext_vector_type(16))) _Float16 v16h;
typedef __attribute__((ext_vector_type(8)))  float    v8f;

// ---------------------------------------------------------------------------
// Kernel 1: fused  s_out = sigmoid(W @ (0.95*s) + 0.05*noise)  AND  W_out = W
// One 16-row block per workgroup; 16 waves each own a 256-wide K slice.
// k-loop unrolled x2 with independent accumulators to kill the WMMA->WMMA
// RAW hazard NOPs; 0.95*s staged once per block in LDS as f16.
// ---------------------------------------------------------------------------
__global__ void __launch_bounds__(512)
matvec_wmma(const float* __restrict__ W, const float* __restrict__ s_in,
            const float* __restrict__ noise, float* __restrict__ s_out,
            float* __restrict__ W_out) {
    const int tid  = threadIdx.x;
    const int wave = tid >> 5;         // 0..15
    const int lane = tid & 31;
    const int half = lane >> 4;        // 0 or 1
    const int row  = blockIdx.x * 16 + (lane & 15);
    const size_t rowoff = (size_t)row * NN;
    const bool col0 = ((lane & 15) == 0);   // lanes feeding column 0 of B

    // Stage 0.95*s as f16 once per block (8 KB LDS), coalesced.
    __shared__ _Float16 sh_s[NN];
    for (int i = tid; i < NN; i += 512) sh_s[i] = (_Float16)(0.95f * s_in[i]);
    __syncthreads();

    v8f c0 = {};
    v8f c1 = {};
    const int kbase = wave * 256;      // this wave's K slice

#pragma unroll 2
    for (int kk = 0; kk < 256; kk += 64) {
        // ---- first 32-wide K step ----
        {
            const int k0   = kbase + kk;
            const int base = k0 + half * 8;     // ISA 16-bit A layout: lanes>=16 shifted by 8
            const float4* p4 = reinterpret_cast<const float4*>(W + rowoff + base);
            float4 x0 = p4[0], x1 = p4[1], x2 = p4[4], x3 = p4[5];
            float4* q4 = reinterpret_cast<float4*>(W_out + rowoff + base);
            q4[0] = x0; q4[1] = x1; q4[4] = x2; q4[5] = x3;

            v16h a;
            a[0]=(_Float16)x0.x;  a[1]=(_Float16)x0.y;  a[2]=(_Float16)x0.z;  a[3]=(_Float16)x0.w;
            a[4]=(_Float16)x1.x;  a[5]=(_Float16)x1.y;  a[6]=(_Float16)x1.z;  a[7]=(_Float16)x1.w;
            a[8]=(_Float16)x2.x;  a[9]=(_Float16)x2.y;  a[10]=(_Float16)x2.z; a[11]=(_Float16)x2.w;
            a[12]=(_Float16)x3.x; a[13]=(_Float16)x3.y; a[14]=(_Float16)x3.z; a[15]=(_Float16)x3.w;

            const int kb = k0 + half * 8;
            v16h b;
#pragma unroll
            for (int e = 0; e < 16; ++e) {
                _Float16 h = sh_s[kb + e + (e >= 8 ? 8 : 0)];       // LDS broadcast read
                b[e] = col0 ? h : (_Float16)0.0f;
            }
            c0 = __builtin_amdgcn_wmma_f32_16x16x32_f16(false, a, false, b,
                                                        (short)0, c0, false, false);
        }
        // ---- second 32-wide K step (independent accumulator) ----
        {
            const int k0   = kbase + kk + 32;
            const int base = k0 + half * 8;
            const float4* p4 = reinterpret_cast<const float4*>(W + rowoff + base);
            float4 x0 = p4[0], x1 = p4[1], x2 = p4[4], x3 = p4[5];
            float4* q4 = reinterpret_cast<float4*>(W_out + rowoff + base);
            q4[0] = x0; q4[1] = x1; q4[4] = x2; q4[5] = x3;

            v16h a;
            a[0]=(_Float16)x0.x;  a[1]=(_Float16)x0.y;  a[2]=(_Float16)x0.z;  a[3]=(_Float16)x0.w;
            a[4]=(_Float16)x1.x;  a[5]=(_Float16)x1.y;  a[6]=(_Float16)x1.z;  a[7]=(_Float16)x1.w;
            a[8]=(_Float16)x2.x;  a[9]=(_Float16)x2.y;  a[10]=(_Float16)x2.z; a[11]=(_Float16)x2.w;
            a[12]=(_Float16)x3.x; a[13]=(_Float16)x3.y; a[14]=(_Float16)x3.z; a[15]=(_Float16)x3.w;

            const int kb = k0 + half * 8;
            v16h b;
#pragma unroll
            for (int e = 0; e < 16; ++e) {
                _Float16 h = sh_s[kb + e + (e >= 8 ? 8 : 0)];
                b[e] = col0 ? h : (_Float16)0.0f;
            }
            c1 = __builtin_amdgcn_wmma_f32_16x16x32_f16(false, a, false, b,
                                                        (short)0, c1, false, false);
        }
    }

    // Column 0 of D lives in lanes 0 (rows 0..7) and 16 (rows 8..15)
    __shared__ float part[16][16];
    if (lane == 0) {
#pragma unroll
        for (int v = 0; v < 8; ++v) part[wave][v] = c0[v] + c1[v];
    } else if (lane == 16) {
#pragma unroll
        for (int v = 0; v < 8; ++v) part[wave][8 + v] = c0[v] + c1[v];
    }
    __syncthreads();
    if (tid < 16) {
        float y = 0.0f;
#pragma unroll
        for (int w = 0; w < 16; ++w) y += part[w][tid];
        const int r = blockIdx.x * 16 + tid;
        const float x = y + 0.05f * noise[r];
        s_out[r] = 1.0f / (1.0f + expf(-x));
    }
}

// ---------------------------------------------------------------------------
// Kernel 2: force young sparks' positions to 1.0 (drop-mode scatter)
// ---------------------------------------------------------------------------
__global__ void force_young(const int* __restrict__ pos, const int* __restrict__ age,
                            float* __restrict__ sd) {
    const int i = threadIdx.x;
    if (i < KSP && age[i] < 5) {
        const int p = pos[i];
        if (p >= 0 && p < NN) sd[p] = 1.0f;
    }
}

// ---------------------------------------------------------------------------
// Kernel 3: sequential 32-spark scan, one 1024-thread workgroup.
// ---------------------------------------------------------------------------
__global__ void __launch_bounds__(1024)
scan_sparks(float* __restrict__ Wd, float* __restrict__ sd,
            const float* __restrict__ energy_in, const float* __restrict__ u_in,
            const int* __restrict__ pos_in, const int* __restrict__ age_in,
            int* __restrict__ pos_out, float* __restrict__ energy_out,
            int* __restrict__ age_out) {
    const int tid = threadIdx.x;
    __shared__ float lds_sum[1024];
    __shared__ float lds_v[1024];
    __shared__ int   lds_i[1024];
    __shared__ int   sh_nxt;
    __shared__ int   sh_top[5];

    for (int i = 0; i < KSP; ++i) {
        const int prev = pos_in[i];
        if (tid == 0) sh_nxt = NN;
        float* Wrow = Wd + (size_t)prev * NN;

        // --- CDF over relu(row)+1e-6, 4 contiguous elements per thread ---
        float4 r4 = *reinterpret_cast<const float4*>(Wrow + 4 * tid);
        float w0 = fmaxf(r4.x, 0.0f) + 1e-6f;
        float w1 = fmaxf(r4.y, 0.0f) + 1e-6f;
        float w2 = fmaxf(r4.z, 0.0f) + 1e-6f;
        float w3 = fmaxf(r4.w, 0.0f) + 1e-6f;
        float lsum = w0 + w1 + w2 + w3;
        lds_sum[tid] = lsum;
        __syncthreads();
        for (int d = 1; d < 1024; d <<= 1) {          // Hillis-Steele inclusive scan
            float t = (tid >= d) ? lds_sum[tid - d] : 0.0f;
            __syncthreads();
            lds_sum[tid] += t;
            __syncthreads();
        }
        const float total = lds_sum[1023];
        const float pre   = lds_sum[tid] - lsum;

        // --- searchsorted(cdf, u*total, 'right'): first index with cdf > t ---
        const float tgt = u_in[i] * total;
        float cacc = pre;
        cacc += w0;
        if (cacc > tgt) atomicMin(&sh_nxt, 4 * tid + 0);
        else { cacc += w1;
            if (cacc > tgt) atomicMin(&sh_nxt, 4 * tid + 1);
            else { cacc += w2;
                if (cacc > tgt) atomicMin(&sh_nxt, 4 * tid + 2);
                else { cacc += w3;
                    if (cacc > tgt) atomicMin(&sh_nxt, 4 * tid + 3);
                }
            }
        }
        __syncthreads();
        const int nxt = min(sh_nxt, NN - 1);

        // --- edge learning on traversed edge ---
        if (tid == 0) {
            const size_t eo = (size_t)nxt * NN + prev;
            const float sp  = sd[prev];
            Wd[eo] = Wd[eo] * 0.95f + sp * 0.05f;
            __threadfence();
        }
        __syncthreads();

        // --- re-read row AFTER edge update; top-5 of relu(row) ---
        float4 t4 = *reinterpret_cast<const float4*>(Wrow + 4 * tid);
        float rv[4] = { fmaxf(t4.x, 0.0f), fmaxf(t4.y, 0.0f),
                        fmaxf(t4.z, 0.0f), fmaxf(t4.w, 0.0f) };
        for (int rpt = 0; rpt < 5; ++rpt) {
            float bv = -1.0f; int bi = 4 * tid;
#pragma unroll
            for (int e = 0; e < 4; ++e)
                if (rv[e] > bv) { bv = rv[e]; bi = 4 * tid + e; }
            lds_v[tid] = bv; lds_i[tid] = bi;
            __syncthreads();
            for (int s2 = 512; s2 > 0; s2 >>= 1) {    // max, tie -> smallest index
                if (tid < s2) {
                    const float ov = lds_v[tid + s2];
                    const int   oi = lds_i[tid + s2];
                    if (ov > lds_v[tid] || (ov == lds_v[tid] && oi < lds_i[tid])) {
                        lds_v[tid] = ov; lds_i[tid] = oi;
                    }
                }
                __syncthreads();
            }
            const int sel = lds_i[0];
            if (tid == 0) sh_top[rpt] = sel;
            if ((sel >> 2) == tid) rv[sel & 3] = -1.0f;   // mask chosen element
            __syncthreads();
        }

        // --- ripple updates + energy/deposit/respawn (serial => deterministic) ---
        if (tid == 0) {
            for (int j = 0; j < 5; ++j) { size_t o = (size_t)prev * NN + sh_top[j];       Wd[o] += 0.01f;  }
            for (int j = 0; j < 5; ++j) { size_t o = (size_t)sh_top[j] * NN + prev;       Wd[o] += 0.005f; }
            for (int j = 0; j < 5; ++j)
                for (int l = 0; l < 5; ++l) { size_t o = (size_t)sh_top[j] * NN + sh_top[l]; Wd[o] += 0.003f; }
            float en = energy_in[i] * 0.98f;
            sd[nxt] = en;                       // deposit BEFORE respawn check
            int ag = age_in[i] + 1;
            int pn = nxt;
            if (en < 0.05f) { pn = i % NN; en = 1.0f; ag = 0; }
            pos_out[i] = pn; energy_out[i] = en; age_out[i] = ag;
            __threadfence();
        }
        __syncthreads();
    }
}

// ---------------------------------------------------------------------------
// Kernel 4: in-place global decay + clamp, float4 streaming
// ---------------------------------------------------------------------------
__global__ void decay_clip(float* __restrict__ Wd) {
    const size_t n4     = (size_t)NN * NN / 4;
    const size_t stride = (size_t)gridDim.x * blockDim.x;
    float4* p = reinterpret_cast<float4*>(Wd);
    for (size_t i = (size_t)blockIdx.x * blockDim.x + threadIdx.x; i < n4; i += stride) {
        float4 v = p[i];
        v.x = fminf(fmaxf(v.x * 0.999f, -2.0f), 2.0f);
        v.y = fminf(fmaxf(v.y * 0.999f, -2.0f), 2.0f);
        v.z = fminf(fmaxf(v.z * 0.999f, -2.0f), 2.0f);
        v.w = fminf(fmaxf(v.w * 0.999f, -2.0f), 2.0f);
        p[i] = v;
    }
}

// ---------------------------------------------------------------------------
// Launch: outputs concatenated as [s(4096), W(4096^2), pos(32,i32), energy(32), age(32,i32)]
// ---------------------------------------------------------------------------
extern "C" void kernel_launch(void* const* d_in, const int* in_sizes, int n_in,
                              void* d_out, int out_size, void* d_ws, size_t ws_size,
                              hipStream_t stream) {
    const float* W            = (const float*)d_in[0];
    const float* s            = (const float*)d_in[1];
    const float* noise        = (const float*)d_in[2];
    const float* spark_energy = (const float*)d_in[3];
    const float* u            = (const float*)d_in[4];
    const int*   spark_pos    = (const int*)d_in[5];
    const int*   spark_age    = (const int*)d_in[6];

    float* out   = (float*)d_out;
    float* s_out = out;                     // [0, 4096)
    float* W_out = out + NN;                // [4096, 4096+N*N)
    const size_t off_pos = (size_t)NN + (size_t)NN * NN;
    int*   pos_out    = (int*)out + off_pos;
    float* energy_out = out + off_pos + KSP;
    int*   age_out    = (int*)out + off_pos + 2 * KSP;

    matvec_wmma<<<dim3(NN / 16), dim3(512), 0, stream>>>(W, s, noise, s_out, W_out);
    force_young<<<dim3(1), dim3(KSP), 0, stream>>>(spark_pos, spark_age, s_out);
    scan_sparks<<<dim3(1), dim3(1024), 0, stream>>>(W_out, s_out, spark_energy, u,
                                                    spark_pos, spark_age,
                                                    pos_out, energy_out, age_out);
    decay_clip<<<dim3(4096), dim3(256), 0, stream>>>(W_out);
}
